// MultiHeadAttention_90159953478259
// MI455X (gfx1250) — matrix operand order repaired and checked by reading, not compiled, not run on hardware
//
#include <hip/hip_runtime.h>
#include <hip/hip_bf16.h>

typedef __attribute__((ext_vector_type(16))) _Float16 v16h;
typedef __attribute__((ext_vector_type(8)))  _Float16 v8h;
typedef __attribute__((ext_vector_type(8)))  float    v8f;

#define HD   384      // per-head dim (== hidden)
#define NH   4
#define BB   16
#define SS   1024
#define M_TOT (BB*SS)         // 16384
#define N_QKV (HD*NH)         // 1536

// ---------------- WMMA helpers (layouts per CDNA5 ISA 7.12.2) ----------------

__device__ inline v8f vzero8() {
  v8f z = {0.f,0.f,0.f,0.f,0.f,0.f,0.f,0.f};
  return z;
}

__device__ inline v8f wmma_f16(v16h a, v16h b, v8f c) {
  return __builtin_amdgcn_wmma_f32_16x16x32_f16(false, a, false, b, (short)0, c, false, false);
}

// A 16x32 f16 tile from row-major [row][k] memory, leading dim ld (elements).
__device__ inline v16h load_a_f16(const _Float16* src, int row0, int ld, int k0, int lane) {
  int m = lane & 15, half = lane >> 4;
  const _Float16* p = src + (size_t)(row0 + m) * ld + k0 + half * 8;
  v8h lo = *(const v8h*)p;
  v8h hi = *(const v8h*)(p + 16);
  v16h r;
#pragma unroll
  for (int i = 0; i < 8; ++i) { r[i] = lo[i]; r[8 + i] = hi[i]; }
  return r;
}

// A 16x32 tile from row-major fp32 memory, converted to f16.
__device__ inline v16h load_a_f32cvt(const float* src, int row0, int ld, int k0, int lane) {
  int m = lane & 15, half = lane >> 4;
  const float* p = src + (size_t)(row0 + m) * ld + k0 + half * 8;
  float4 x0 = *(const float4*)(p);
  float4 x1 = *(const float4*)(p + 4);
  float4 y0 = *(const float4*)(p + 16);
  float4 y1 = *(const float4*)(p + 20);
  v16h r;
  r[0]=(_Float16)x0.x;  r[1]=(_Float16)x0.y;  r[2]=(_Float16)x0.z;  r[3]=(_Float16)x0.w;
  r[4]=(_Float16)x1.x;  r[5]=(_Float16)x1.y;  r[6]=(_Float16)x1.z;  r[7]=(_Float16)x1.w;
  r[8]=(_Float16)y0.x;  r[9]=(_Float16)y0.y;  r[10]=(_Float16)y0.z; r[11]=(_Float16)y0.w;
  r[12]=(_Float16)y1.x; r[13]=(_Float16)y1.y; r[14]=(_Float16)y1.z; r[15]=(_Float16)y1.w;
  return r;
}

// B 32x16 f16 tile from N-major memory ([n][k] row-major), leading dim ld.
__device__ inline v16h load_b_f16(const _Float16* src, int n0, int ld, int k0, int lane) {
  int n = lane & 15, half = lane >> 4;
  const _Float16* p = src + (size_t)(n0 + n) * ld + k0 + half * 8;
  v8h lo = *(const v8h*)p;
  v8h hi = *(const v8h*)(p + 16);
  v16h r;
#pragma unroll
  for (int i = 0; i < 8; ++i) { r[i] = lo[i]; r[8 + i] = hi[i]; }
  return r;
}

// ---------------- prep kernels ----------------

__global__ void cvt_f32_to_f16(const float* __restrict__ in, _Float16* __restrict__ out, int n) {
  int i = blockIdx.x * blockDim.x + threadIdx.x;
  if (i < n) out[i] = (_Float16)in[i];
}

// W [K][N] fp32 row-major  ->  WT [N][K] f16 row-major
__global__ void transpose_cvt(const float* __restrict__ W, _Float16* __restrict__ WT, int K, int N) {
  int i = blockIdx.x * blockDim.x + threadIdx.x;
  if (i >= K * N) return;
  int k = i / N, n = i % N;
  WT[(size_t)n * K + k] = (_Float16)W[(size_t)k * N + n];
}

// ---------------- QKV projection: C = X @ W + b ----------------
// X f16 [16384][384], WT f16 [1536][384] (N-major), bias fp32 [1536].
// mode 0: out[head][m][h]  (q/k head-major, seq-row-major)
// mode 1: out[head][b][h][s] (v transposed so attn@V B-tiles are contiguous)
__global__ void qkv_gemm(const _Float16* __restrict__ X, const _Float16* __restrict__ WT,
                         const float* __restrict__ bias, _Float16* __restrict__ out, int mode) {
  int nt = blockIdx.x, mt = blockIdx.y, lane = threadIdx.x;
  v8f acc = vzero8();
#pragma unroll
  for (int k0 = 0; k0 < HD; k0 += 32) {
    v16h a = load_a_f16(X, mt * 16, HD, k0, lane);
    v16h b = load_b_f16(WT, nt * 16, HD, k0, lane);
    acc = wmma_f16(a, b, acc);
  }
  int n = nt * 16 + (lane & 15);
  int head = n / HD, h = n % HD;
  float bv = bias[n];
#pragma unroll
  for (int i = 0; i < 8; ++i) {
    int m = mt * 16 + i + (lane >> 4) * 8;
    float v = acc[i] + bv;
    if (mode == 0) {
      out[((size_t)head * M_TOT + m) * HD + h] = (_Float16)v;
    } else {
      int b_ = m >> 10, s = m & 1023;
      out[(((size_t)head * BB + b_) * HD + h) * SS + s] = (_Float16)v;
    }
  }
}

// ---------------- scores = (qh @ khT) * scale, mask, softmax -> attn (fp32) ----------------
// one wave per (head, b, 16-query-row block); 16x1024 fp32 scores staged in 64KB LDS.
__global__ void scores_softmax(const _Float16* __restrict__ qh, const _Float16* __restrict__ kh,
                               const unsigned char* __restrict__ mask, float* __restrict__ attn) {
  __shared__ float sc[16 * SS];   // 64 KB of the 320 KB WGP LDS
  int qt = blockIdx.x, bb = blockIdx.y, head = blockIdx.z, lane = threadIdx.x;
  const float scale = 0.051031036307982884f;  // 1/sqrt(384)

  const _Float16* qH = qh + (size_t)head * M_TOT * HD;
  const _Float16* kH = kh + (size_t)head * M_TOT * HD;
  int qrow0 = bb * SS + qt * 16;
  int krow0 = bb * SS;

  v16h a[12];
#pragma unroll
  for (int kc = 0; kc < 12; ++kc) a[kc] = load_a_f16(qH, qrow0, HD, kc * 32, lane);

  for (int kt = 0; kt < 64; ++kt) {
    v8f acc = vzero8();
#pragma unroll
    for (int kc = 0; kc < 12; ++kc) {
      v16h b = load_b_f16(kH, krow0 + kt * 16, HD, kc * 32, lane);
      acc = wmma_f16(a[kc], b, acc);
    }
    int n = kt * 16 + (lane & 15);
#pragma unroll
    for (int i = 0; i < 8; ++i) {
      int r = i + (lane >> 4) * 8;
      sc[r * SS + n] = acc[i] * scale;
    }
  }
  __syncthreads();

  const unsigned char* mrow = mask + ((size_t)bb * SS + qt * 16) * SS;
  for (int r = 0; r < 16; ++r) {
    int q = qt * 16 + r;
    float mx = -1e30f;
    for (int c = lane; c < SS; c += 32) {
      float v = sc[r * SS + c];
      v = mrow[(size_t)r * SS + c] ? v : -1e9f;
      sc[r * SS + c] = v;
      mx = fmaxf(mx, v);
    }
#pragma unroll
    for (int o = 16; o > 0; o >>= 1) mx = fmaxf(mx, __shfl_xor(mx, o, 32));
    float sum = 0.f;
    for (int c = lane; c < SS; c += 32) {
      float e = __expf(sc[r * SS + c] - mx);
      sc[r * SS + c] = e;
      sum += e;
    }
#pragma unroll
    for (int o = 16; o > 0; o >>= 1) sum += __shfl_xor(sum, o, 32);
    float inv = 1.0f / sum;
    float* arow = attn + (((size_t)head * BB + bb) * SS + q) * SS;
    for (int c = lane; c < SS; c += 32) arow[c] = sc[r * SS + c] * inv;
  }
}

// ---------------- ctx = attn @ V (merged heads, f16) ----------------
__global__ void attn_v(const float* __restrict__ attn, const _Float16* __restrict__ vhT,
                       _Float16* __restrict__ ctx) {
  int lane = threadIdx.x;
  int hb = blockIdx.z, head = hb >> 4, bb = hb & 15;
  const float*    A  = attn + ((size_t)head * BB + bb) * SS * SS;   // [S][S]
  const _Float16* Bv = vhT  + ((size_t)head * BB + bb) * HD * SS;   // [h][s], ld = S
  v8f acc = vzero8();
#pragma unroll 4
  for (int k0 = 0; k0 < SS; k0 += 32) {
    v16h a = load_a_f32cvt(A, blockIdx.y * 16, SS, k0, lane);
    v16h b = load_b_f16(Bv, blockIdx.x * 16, SS, k0, lane);
    acc = wmma_f16(a, b, acc);
  }
  int h = blockIdx.x * 16 + (lane & 15);
#pragma unroll
  for (int i = 0; i < 8; ++i) {
    int s = blockIdx.y * 16 + i + (lane >> 4) * 8;
    ctx[((size_t)bb * SS + s) * N_QKV + head * HD + h] = (_Float16)acc[i];
  }
}

// ---------------- out-proj + residual + LayerNorm ----------------
// one wave per 16-row block: 24 resident accumulators (16x384 output), LDS row staging.
__global__ void proj_ln(const _Float16* __restrict__ ctx, const _Float16* __restrict__ WpT,
                        const float* __restrict__ bp, const float* __restrict__ q,
                        const float* __restrict__ gamma, const float* __restrict__ beta,
                        float* __restrict__ out) {
  __shared__ float sc[16 * HD];   // 24 KB
  int mt = blockIdx.x, lane = threadIdx.x;

  v8f acc[24];
#pragma unroll
  for (int t = 0; t < 24; ++t) acc[t] = vzero8();

  for (int k0 = 0; k0 < N_QKV; k0 += 32) {
    v16h a = load_a_f16(ctx, mt * 16, N_QKV, k0, lane);
#pragma unroll
    for (int t = 0; t < 24; ++t) {
      v16h b = load_b_f16(WpT, t * 16, N_QKV, k0, lane);
      acc[t] = wmma_f16(a, b, acc[t]);
    }
  }

#pragma unroll
  for (int t = 0; t < 24; ++t) {
    int n = t * 16 + (lane & 15);
    float bv = bp[n];
#pragma unroll
    for (int i = 0; i < 8; ++i) {
      int r = i + (lane >> 4) * 8;
      sc[r * HD + n] = acc[t][i] + bv;
    }
  }
  __syncthreads();

  for (int r = 0; r < 16; ++r) {
    int m = mt * 16 + r;
    const float* qrow = q + (size_t)m * HD;
    float sum = 0.f;
    for (int c = lane; c < HD; c += 32) {
      float v = sc[r * HD + c] + qrow[c];
      sc[r * HD + c] = v;
      sum += v;
    }
#pragma unroll
    for (int o = 16; o > 0; o >>= 1) sum += __shfl_xor(sum, o, 32);
    float mean = sum * (1.0f / HD);
    float var = 0.f;
    for (int c = lane; c < HD; c += 32) {
      float d = sc[r * HD + c] - mean;
      var += d * d;
    }
#pragma unroll
    for (int o = 16; o > 0; o >>= 1) var += __shfl_xor(var, o, 32);
    var *= (1.0f / HD);
    float inv = rsqrtf(var + 1e-5f);
    float* orow = out + (size_t)m * HD;
    for (int c = lane; c < HD; c += 32)
      orow[c] = (sc[r * HD + c] - mean) * inv * gamma[c] + beta[c];
  }
}

// ---------------- launch ----------------

extern "C" void kernel_launch(void* const* d_in, const int* in_sizes, int n_in,
                              void* d_out, int out_size, void* d_ws, size_t ws_size,
                              hipStream_t stream) {
  const float* q    = (const float*)d_in[0];
  const unsigned char* mask = (const unsigned char*)d_in[1];
  const float* Wq   = (const float*)d_in[2];
  const float* bq   = (const float*)d_in[3];
  const float* Wk   = (const float*)d_in[4];
  const float* bk   = (const float*)d_in[5];
  const float* Wv   = (const float*)d_in[6];
  const float* bv   = (const float*)d_in[7];
  const float* Wp   = (const float*)d_in[8];
  const float* bp   = (const float*)d_in[9];
  const float* gamma= (const float*)d_in[10];
  const float* beta = (const float*)d_in[11];

  char* ws = (char*)d_ws;
  size_t off = 0;
  _Float16* Xf  = (_Float16*)(ws + off); off += (size_t)M_TOT * HD * 2;       // 12.6 MB
  _Float16* WqT = (_Float16*)(ws + off); off += (size_t)N_QKV * HD * 2;       // 1.18 MB
  _Float16* WkT = (_Float16*)(ws + off); off += (size_t)N_QKV * HD * 2;
  _Float16* WvT = (_Float16*)(ws + off); off += (size_t)N_QKV * HD * 2;
  _Float16* WpT = (_Float16*)(ws + off); off += (size_t)HD * N_QKV * 2;
  _Float16* qh  = (_Float16*)(ws + off); off += (size_t)NH * M_TOT * HD * 2;  // 50 MB
  _Float16* kh  = (_Float16*)(ws + off); off += (size_t)NH * M_TOT * HD * 2;  // 50 MB
  _Float16* vhT = (_Float16*)(ws + off); off += (size_t)NH * M_TOT * HD * 2;  // 50 MB
  _Float16* ctx = (_Float16*)(ws + off); off += (size_t)M_TOT * N_QKV * 2;    // 50 MB

  float* result = (float*)d_out;                      // [16,1024,384]
  float* attn   = result + (size_t)M_TOT * HD;        // [64,1024,1024]

  // 1) precision conversion / weight transposes
  cvt_f32_to_f16<<<(M_TOT * HD + 255) / 256, 256, 0, stream>>>(q, Xf, M_TOT * HD);
  transpose_cvt<<<(HD * N_QKV + 255) / 256, 256, 0, stream>>>(Wq, WqT, HD, N_QKV);
  transpose_cvt<<<(HD * N_QKV + 255) / 256, 256, 0, stream>>>(Wk, WkT, HD, N_QKV);
  transpose_cvt<<<(HD * N_QKV + 255) / 256, 256, 0, stream>>>(Wv, WvT, HD, N_QKV);
  transpose_cvt<<<(N_QKV * HD + 255) / 256, 256, 0, stream>>>(Wp, WpT, N_QKV, HD);

  // 2) QKV projections
  qkv_gemm<<<dim3(N_QKV / 16, M_TOT / 16), 32, 0, stream>>>(Xf, WqT, bq, qh, 0);
  qkv_gemm<<<dim3(N_QKV / 16, M_TOT / 16), 32, 0, stream>>>(Xf, WkT, bk, kh, 0);
  qkv_gemm<<<dim3(N_QKV / 16, M_TOT / 16), 32, 0, stream>>>(Xf, WvT, bv, vhT, 1);

  // 3) scores + softmax -> attn (fp32 output section of d_out)
  scores_softmax<<<dim3(SS / 16, BB, NH), 32, 0, stream>>>(qh, kh, mask, attn);

  // 4) ctx = attn @ V (heads merged)
  attn_v<<<dim3(HD / 16, SS / 16, NH * BB), 32, 0, stream>>>(attn, vhT, ctx);

  // 5) output projection + residual + LayerNorm
  proj_ln<<<M_TOT / 16, 32, 0, stream>>>(ctx, WpT, bp, q, gamma, beta, result);
}
